// RelationExtractorCnnPosNetwork_60808146977249
// MI455X (gfx1250) — matrix operand order repaired and checked by reading, not compiled, not run on hardware
//
#include <hip/hip_runtime.h>
#include <hip/hip_bf16.h>
#include <stdint.h>

// ---------------- problem constants (from reference) ----------------
#define BB       256
#define SS       512
#define EMBD     300
#define POSD     16
#define C_IN     332          // EMB_D + 2*POS_D
#define CP       352          // channels padded to multiple of 32
#define CNN_OUT  50
#define KW       5
#define MP       64           // M padded (50 -> 64)
#define KP       (CP * KW)    // 1760, GEMM K dim
#define NCLASS   10
#define SBLK     64           // output columns per block
#define JCOLS    (SBLK + 4)   // +halo (2 each side)

typedef __attribute__((ext_vector_type(8)))  float  v8f;
typedef __attribute__((ext_vector_type(16))) __bf16 v16bf;
typedef __attribute__((ext_vector_type(8)))  __bf16 v8bf;

// ---------------- zero the max-pool accumulator ----------------
__global__ void init_ymax_kernel(float* __restrict__ ymax) {
    int t = blockIdx.x * blockDim.x + threadIdx.x;
    if (t < BB * MP) ymax[t] = 0.0f;
}

// ---------------- prepack conv weights: fp32 [50,332,5] -> bf16 Wp[64][1760]
// K-dim layout: kk = k*CP + c  (each 32-wide chunk stays within one tap k)
__global__ void prepack_w_kernel(const float* __restrict__ convw,
                                 __bf16* __restrict__ Wp) {
    int t = blockIdx.x * blockDim.x + threadIdx.x;
    if (t >= MP * KP) return;
    int m  = t / KP;
    int kk = t % KP;
    int k  = kk / CP;
    int c  = kk % CP;
    float v = 0.0f;
    if (m < CNN_OUT && c < C_IN)
        v = convw[(m * C_IN + c) * KW + k];
    Wp[t] = (__bf16)v;
}

// ---------------- fused gather + im2col(LDS) + WMMA conv + relu + maxpool ----
__global__ __launch_bounds__(256)
void conv_wmma_kernel(const int*   __restrict__ tokens,
                      const float* __restrict__ embed,
                      const float* __restrict__ posT,
                      const float* __restrict__ convb,
                      const __bf16* __restrict__ Wp,
                      const int* __restrict__ e1p,
                      const int* __restrict__ e2p,
                      float* __restrict__ ymax) {
    const int b    = blockIdx.y;
    const int s0   = blockIdx.x * SBLK;
    const int tid  = threadIdx.x;
    const int lane = tid & 31;
    const int wave = tid >> 5;

    __shared__ int   tokLds[SS];
    __shared__ short dLds[JCOLS][2];
    __shared__ __align__(32) __bf16 Xs[JCOLS * CP];

    // --- stage token row via CDNA5 async global->LDS copy (ASYNCcnt path) ---
    {
        unsigned ldsOff =
            (unsigned)(uintptr_t)(&tokLds[0]) + (unsigned)(tid * 8);
        unsigned long long gaddr =
            (unsigned long long)(uintptr_t)(tokens + b * SS) +
            (unsigned long long)(tid * 8);
        asm volatile("global_load_async_to_lds_b64 %0, %1, off"
                     :: "v"(ldsOff), "v"(gaddr) : "memory");
        asm volatile("s_wait_asynccnt 0x0" ::: "memory");
    }
    __syncthreads();

    const int e1 = *e1p, e2 = *e2p;

    // --- nearest-entity distances (per column, per entity) ---
    for (int t = tid; t < JCOLS * 2; t += 256) {
        int j   = t >> 1;
        int ent = (t & 1) ? e2 : e1;
        int s   = s0 + j - 2;
        int best = 0;
        if (s >= 0 && s < SS) {
            best = SS - 1;   // clip(dist, 0, POS_MAX-1), no-occurrence -> max
            for (int sp = 0; sp < SS; ++sp) {
                if (tokLds[sp] == ent) {
                    int d = (s > sp) ? (s - sp) : (sp - s);
                    best = (d < best) ? d : best;
                }
            }
        }
        dLds[j][t & 1] = (short)best;
    }
    __syncthreads();

    // --- gather features into LDS tile (one wave per column) ---
    for (int j = wave; j < JCOLS; j += 8) {
        int  s     = s0 + j - 2;
        bool valid = (s >= 0 && s < SS);
        int  tok   = valid ? tokLds[s] : 0;
        const float* er = embed + (long)tok * EMBD;

        unsigned nz = 0u;
        for (int c = lane; c < EMBD; c += 32) {
            float f = valid ? er[c] : 0.0f;
            nz |= (f != 0.0f) ? 1u : 0u;
            Xs[j * CP + c] = (__bf16)f;
        }
        bool pad = (__ballot(nz != 0u) == 0ull);   // zero embedding row

        // positional channels 300..331
        {
            int d1 = dLds[j][0], d2 = dLds[j][1];
            float f;
            if (lane < 16) f = (valid && !pad) ? posT[d1 * POSD + lane]        : 0.0f;
            else           f = (valid && !pad) ? posT[d2 * POSD + (lane - 16)] : 0.0f;
            Xs[j * CP + EMBD + lane] = (__bf16)f;
        }
        // channel padding 332..351
        for (int c = C_IN + lane; c < CP; c += 32)
            Xs[j * CP + c] = (__bf16)0.0f;
    }
    __syncthreads();

    // --- WMMA implicit GEMM: wave -> ntile = wave>>1, mtiles {w&1, (w&1)+2}
    const int ntile = wave >> 1;        // 0..3 (16 output cols each)
    const int mbase = wave & 1;         // mtiles mbase, mbase+2
    const int n     = lane & 15;        // N index (and A's M index)
    const int half  = lane >> 4;
    const int ksA   = half * 8;         // A frag: K runs at ksA and ksA+16
    const int kbB   = half * 8;         // B frag: K runs at kbB and kbB+16

    v8f acc0 = {};
    v8f acc1 = {};

    // base pointers; inner-loop addresses are base + constant immediates
    const __bf16* aBase0 = Wp + (long)(mbase * 16 + n) * KP + ksA;       // mtile mbase
    const __bf16* aBase1 = Wp + (long)((mbase + 2) * 16 + n) * KP + ksA; // mtile mbase+2

    #pragma unroll 1
    for (int k = 0; k < KW; ++k) {
        const __bf16* xrow = &Xs[(ntile * 16 + n + k) * CP + kbB];
        const __bf16* a0   = aBase0 + k * CP;
        const __bf16* a1   = aBase1 + k * CP;
        #pragma unroll
        for (int cc = 0; cc < CP / 32; ++cc) {
            v16bf bfrag;
            {
                v8bf lo = *(const v8bf*)(xrow + cc * 32);
                v8bf hi = *(const v8bf*)(xrow + cc * 32 + 16);
                __builtin_memcpy(&bfrag, &lo, 16);
                __builtin_memcpy((char*)&bfrag + 16, &hi, 16);
            }

            v16bf af0, af1;
            {
                v8bf lo = *(const v8bf*)(a0 + cc * 32);
                v8bf hi = *(const v8bf*)(a0 + cc * 32 + 16);
                __builtin_memcpy(&af0, &lo, 16);
                __builtin_memcpy((char*)&af0 + 16, &hi, 16);
            }
            {
                v8bf lo = *(const v8bf*)(a1 + cc * 32);
                v8bf hi = *(const v8bf*)(a1 + cc * 32 + 16);
                __builtin_memcpy(&af1, &lo, 16);
                __builtin_memcpy((char*)&af1 + 16, &hi, 16);
            }

            acc0 = __builtin_amdgcn_wmma_f32_16x16x32_bf16(
                false, af0, false, bfrag, (short)0, acc0, false, false);
            acc1 = __builtin_amdgcn_wmma_f32_16x16x32_bf16(
                false, af1, false, bfrag, (short)0, acc1, false, false);
        }
    }

    // --- epilogue: bias + relu + max over the 16 columns of the tile,
    //     then non-negative-float atomicMax (int-bits ordering) ---
    #pragma unroll
    for (int which = 0; which < 2; ++which) {
        v8f acc    = which ? acc1 : acc0;
        int mtile  = which ? (mbase + 2) : mbase;
        #pragma unroll
        for (int r = 0; r < 8; ++r) {
            int m   = mtile * 16 + r + 8 * half;   // C/D layout: M = r + 8*half
            float v = acc[r];
            if (m < CNN_OUT) v += convb[m];
            v = fmaxf(v, 0.0f);
            #pragma unroll
            for (int off = 1; off < 16; off <<= 1)
                v = fmaxf(v, __shfl_xor(v, off, 32));
            if (n == 0 && m < CNN_OUT)
                atomicMax((int*)&ymax[b * MP + m], __float_as_int(v));
        }
    }
}

// ---------------- tiny FC: [256,50] @ [50,10]^T + b ----------------
__global__ void fc_kernel(const float* __restrict__ ymax,
                          const float* __restrict__ fcw,
                          const float* __restrict__ fcb,
                          float* __restrict__ out) {
    int t = blockIdx.x * blockDim.x + threadIdx.x;
    if (t >= BB * NCLASS) return;
    int b  = t / NCLASS;
    int nc = t % NCLASS;
    float s = fcb[nc];
    for (int o = 0; o < CNN_OUT; ++o)
        s += ymax[b * MP + o] * fcw[nc * CNN_OUT + o];
    out[t] = s;
}

extern "C" void kernel_launch(void* const* d_in, const int* in_sizes, int n_in,
                              void* d_out, int out_size, void* d_ws, size_t ws_size,
                              hipStream_t stream) {
    (void)in_sizes; (void)n_in; (void)out_size; (void)ws_size;

    const int*   tokens = (const int*)  d_in[0];
    const float* embed  = (const float*)d_in[1];
    const float* posT   = (const float*)d_in[2];
    const float* convw  = (const float*)d_in[3];
    const float* convb  = (const float*)d_in[4];
    const float* fcw    = (const float*)d_in[5];
    const float* fcb    = (const float*)d_in[6];
    const int*   e1     = (const int*)  d_in[7];
    const int*   e2     = (const int*)  d_in[8];

    __bf16* Wp   = (__bf16*)d_ws;                                  // 64*1760*2 = 225,280 B
    float*  ymax = (float*)((char*)d_ws + (size_t)MP * KP * 2);    // 256*64*4  =  65,536 B
    float*  out  = (float*)d_out;

    init_ymax_kernel<<<(BB * MP + 255) / 256, 256, 0, stream>>>(ymax);
    prepack_w_kernel<<<(MP * KP + 255) / 256, 256, 0, stream>>>(convw, Wp);

    dim3 grid(SS / SBLK, BB);   // 8 x 256 blocks
    conv_wmma_kernel<<<grid, 256, 0, stream>>>(tokens, embed, posT, convb,
                                               Wp, e1, e2, ymax);

    fc_kernel<<<(BB * NCLASS + 255) / 256, 256, 0, stream>>>(ymax, fcw, fcb, out);
}
